// peepLSTM_13812614824180
// MI455X (gfx1250) — matrix operand order repaired and checked by reading, not compiled, not run on hardware
//
#include <hip/hip_runtime.h>

#define B_ 256
#define T_ 512
#define D_ 128
#define H_ 512
#define C_ 10

typedef float v8f __attribute__((ext_vector_type(8)));
typedef __bf16 bf16x8 __attribute__((ext_vector_type(8)));
typedef __bf16 bf16x16 __attribute__((ext_vector_type(16)));

__device__ __forceinline__ unsigned short f2bf(float f) {
  unsigned u = __float_as_uint(f);
  u += 0x7fffu + ((u >> 16) & 1u);   // round-to-nearest-even
  return (unsigned short)(u >> 16);
}
__device__ __forceinline__ float sigm(float x) { return 1.0f / (1.0f + __expf(-x)); }
__device__ __forceinline__ float tanh_fast(float x) {
  float e = __expf(2.0f * x);
  return (e - 1.0f) / (e + 1.0f);
}

// ---------------------------------------------------------------------------
// Pre-pass 1: convert the 3 recurrent weights (f32, row-major [H][H]) to bf16.
// Row-major bf16 is already the correct feed for the WMMA B operand:
// lane<16 reads W[n][k0..k0+16), lane>=16 reads W[n][k0+16..k0+32).
// ---------------------------------------------------------------------------
__global__ void cvt_weights_kernel(const float* __restrict__ Wf,
                                   const float* __restrict__ Wi,
                                   const float* __restrict__ Wo,
                                   unsigned short* __restrict__ out) {
  int i = blockIdx.x * blockDim.x + threadIdx.x;          // 0 .. 3*H*H-1
  const float* src = (i < H_ * H_) ? Wf : (i < 2 * H_ * H_ ? Wi : Wo);
  int idx = i & (H_ * H_ - 1);
  out[i] = f2bf(src[idx]);
}

// ---------------------------------------------------------------------------
// Pre-pass 2: the input-side gates depend only on the 3-symbol alphabet:
// gate_pre[sym][gate][h] = emb[sym] . W*x[h] + b*  (3*4*512 = 24 KB table)
// This replaces the reference's 69-GFLOP / 1-GB "gates" tensor.
// The gc gate (g==3) has no recurrent term, so we bake sigmoid() into the
// table and delete one exp chain per element from the serial recurrence.
// ---------------------------------------------------------------------------
__global__ void gate_pre_kernel(const float* __restrict__ emb,
                                const float* __restrict__ Wfx, const float* __restrict__ bf,
                                const float* __restrict__ Wix, const float* __restrict__ bi,
                                const float* __restrict__ Wox, const float* __restrict__ bo,
                                const float* __restrict__ Wcx, const float* __restrict__ bc,
                                float* __restrict__ gpre) {
  int i = blockIdx.x * blockDim.x + threadIdx.x;          // 0 .. 3*4*512-1
  int h = i & (H_ - 1);
  int g = (i >> 9) & 3;
  int s = i >> 11;
  const float* W  = (g == 0) ? Wfx : (g == 1) ? Wix : (g == 2) ? Wox : Wcx;
  const float* bb = (g == 0) ? bf  : (g == 1) ? bi  : (g == 2) ? bo  : bc;
  const float* e = emb + s * D_;
  const float* w = W + h * D_;
  float acc = bb[h];
#pragma unroll 8
  for (int d = 0; d < D_; ++d) acc += e[d] * w[d];
  gpre[i] = (g == 3) ? sigm(acc) : acc;                   // layout [s][g][h]
}

// ---------------------------------------------------------------------------
// Main persistent recurrence kernel. One WG owns 16 batch rows for all T=512
// steps (rows of B are independent in the scan). 16 waves (512 thr) x 2
// N-tiles cover H=512; 4 waves/SIMD so one wave's L2 weight-load latency
// hides under other waves' WMMAs. Per step: 16 k-steps x 6 WMMA per wave
// (3 gates x 2 tiles) in bf16. c lives in registers (f32); the bf16 copy
// round-trips LDS for the shared A operand.
// ---------------------------------------------------------------------------
__global__ void __launch_bounds__(512, 1)
lstm_recurrence_kernel(const int* __restrict__ x,
                       const unsigned short* __restrict__ Wbf,   // [3][H][H] bf16
                       const float* __restrict__ gpre_g,         // [3][4][H] f32
                       const float* __restrict__ Wph,            // [C][H] f32
                       const float* __restrict__ bp,             // [C]
                       float* __restrict__ out) {                // [B][C]
  extern __shared__ unsigned char smem[];
  unsigned short* cbf = (unsigned short*)smem;                     //  0      .. 16384  (16x512 bf16)
  float* gpre = (float*)(smem + 16384);                            //  16384  .. 40960  (3x4x512 f32)
  int*   xs   = (int*)(smem + 40960);                              //  40960  .. 41024  (+pad)
  float* hshr = (float*)(smem + 41088);                            //  41088  .. 73856  (16x512 f32)
  float* pshr = (float*)(smem + 73856);                            //  73856  .. 74496  (16x10 f32)

  const int tid   = threadIdx.x;
  const int lane  = tid & 31;
  const int wave  = tid >> 5;             // 0..15
  const int laneN = lane & 15;
  const int hi    = lane >> 4;            // 0: lanes 0-15, 1: lanes 16-31
  const int b0    = blockIdx.x * 16;
  const int n0w   = wave * 32;            // 2 N-tiles of 16 per wave

  const unsigned short* Wf = Wbf;
  const unsigned short* Wi = Wbf + H_ * H_;
  const unsigned short* Wo = Wbf + 2 * H_ * H_;

  // hint: pull this wave's weight stripe toward the caches
  __builtin_prefetch(Wf + n0w * H_, 0, 1);
  __builtin_prefetch(Wi + n0w * H_, 0, 1);
  __builtin_prefetch(Wo + n0w * H_, 0, 1);

  // stage gate table into LDS, zero c_bf16, load first step's symbols
  for (int i = tid; i < 3 * 4 * H_; i += 512) gpre[i] = gpre_g[i];
  for (int i = tid; i < 16 * H_; i += 512) cbf[i] = 0;
  if (tid < 16) xs[tid] = x[(b0 + tid) * T_ + 0];
  __syncthreads();

  float c_reg[2][8];
#pragma unroll
  for (int ti = 0; ti < 2; ++ti)
#pragma unroll
    for (int j = 0; j < 8; ++j) c_reg[ti][j] = 0.0f;

  for (int t = 0; t < T_; ++t) {
    v8f acc[3][2];
#pragma unroll
    for (int g = 0; g < 3; ++g)
#pragma unroll
      for (int ti = 0; ti < 2; ++ti)
#pragma unroll
        for (int j = 0; j < 8; ++j) acc[g][ti][j] = 0.0f;

    // ---- 16x512 @ 512x512 (x3 gates) : K blocked by 32 -------------------
    for (int kt = 0; kt < H_ / 32; ++kt) {
      const int k0 = kt * 32;
      const int aoff = hi ? 16 : 0;
      // A operand: c row (m = laneN), contiguous 16 k per lane half (k = 16h + s)
      const bf16x8 alo = *(const bf16x8*)(cbf + laneN * H_ + k0 + aoff);
      const bf16x8 ahi = *(const bf16x8*)(cbf + laneN * H_ + k0 + 8 + aoff);
      const bf16x16 a = __builtin_shufflevector(alo, ahi,
          0, 1, 2, 3, 4, 5, 6, 7, 8, 9, 10, 11, 12, 13, 14, 15);
      const int kb = k0 + (hi ? 16 : 0);
#pragma unroll
      for (int ti = 0; ti < 2; ++ti) {
        const int nrow = n0w + ti * 16 + laneN;
        const bf16x16 bF = *(const bf16x16*)(Wf + nrow * H_ + kb);
        acc[0][ti] = __builtin_amdgcn_wmma_f32_16x16x32_bf16(
            false, a, false, bF, (short)0, acc[0][ti], false, false);
        const bf16x16 bI = *(const bf16x16*)(Wi + nrow * H_ + kb);
        acc[1][ti] = __builtin_amdgcn_wmma_f32_16x16x32_bf16(
            false, a, false, bI, (short)0, acc[1][ti], false, false);
        const bf16x16 bO = *(const bf16x16*)(Wo + nrow * H_ + kb);
        acc[2][ti] = __builtin_amdgcn_wmma_f32_16x16x32_bf16(
            false, a, false, bO, (short)0, acc[2][ti], false, false);
      }
    }

    // ---- gate nonlinearity + c update (lane owns exactly its acc cells) --
#pragma unroll
    for (int ti = 0; ti < 2; ++ti) {
      const int n = n0w + ti * 16 + laneN;
#pragma unroll
      for (int j = 0; j < 8; ++j) {
        const int m = j + (hi ? 8 : 0);
        const int sym = xs[m];
        const float* gp = gpre + sym * 4 * H_;
        const float gf = acc[0][ti][j] + gp[0 * H_ + n];
        const float gi = acc[1][ti][j] + gp[1 * H_ + n];
        const float go = acc[2][ti][j] + gp[2 * H_ + n];
        const float sgc = gp[3 * H_ + n];            // sigmoid pre-applied in table
        const float f  = sigm(gf);
        const float ii = sigm(gi);
        const float o  = sigm(go);
        const float cn = sgc * ii + c_reg[ti][j] * f;
        c_reg[ti][j] = cn;
        cbf[m * H_ + n] = f2bf(cn);
        if (t == T_ - 1) hshr[m * H_ + n] = tanh_fast(cn) * o;  // h only needed at t=T-1
      }
    }
    __syncthreads();                       // cbf + xs reads done
    if (tid < 16 && t + 1 < T_) xs[tid] = x[(b0 + tid) * T_ + (t + 1)];
    __syncthreads();                       // new xs visible
  }

  // ---- epilogue: p = h @ Wph^T + bp ; log_softmax ------------------------
  if (tid < 16 * C_) {
    const int m = tid / C_, c = tid % C_;
    float s = bp[c];
    const float* w  = Wph + c * H_;
    const float* hh = hshr + m * H_;
#pragma unroll 8
    for (int k = 0; k < H_; ++k) s += hh[k] * w[k];
    pshr[tid] = s;
  }
  __syncthreads();
  if (tid < 16) {
    float mx = -1e30f;
    for (int c = 0; c < C_; ++c) mx = fmaxf(mx, pshr[tid * C_ + c]);
    float se = 0.0f;
    for (int c = 0; c < C_; ++c) se += __expf(pshr[tid * C_ + c] - mx);
    const float lse = mx + __logf(se);
    for (int c = 0; c < C_; ++c)
      out[(b0 + tid) * C_ + c] = pshr[tid * C_ + c] - lse;
  }
}

// ---------------------------------------------------------------------------
extern "C" void kernel_launch(void* const* d_in, const int* in_sizes, int n_in,
                              void* d_out, int out_size, void* d_ws, size_t ws_size,
                              hipStream_t stream) {
  const int*   x   = (const int*)d_in[0];
  const float* emb = (const float*)d_in[1];
  const float* Wfx = (const float*)d_in[2];
  const float* Wfh = (const float*)d_in[3];
  const float* bf  = (const float*)d_in[4];
  const float* Wix = (const float*)d_in[5];
  const float* Wih = (const float*)d_in[6];
  const float* bi  = (const float*)d_in[7];
  const float* Wox = (const float*)d_in[8];
  const float* Woh = (const float*)d_in[9];
  const float* bo  = (const float*)d_in[10];
  const float* Wcx = (const float*)d_in[11];
  const float* bc  = (const float*)d_in[12];
  const float* Wph = (const float*)d_in[13];
  const float* bp  = (const float*)d_in[14];

  unsigned short* Wbf = (unsigned short*)d_ws;                       // 3*H*H bf16 = 1.5 MB
  float* gpre = (float*)((char*)d_ws + (size_t)3 * H_ * H_ * 2);     // 24 KB

  cvt_weights_kernel<<<(3 * H_ * H_) / 256, 256, 0, stream>>>(Wfh, Wih, Woh, Wbf);
  gate_pre_kernel<<<(3 * 4 * H_) / 256, 256, 0, stream>>>(
      emb, Wfx, bf, Wix, bi, Wox, bo, Wcx, bc, gpre);

  const int smem_bytes = 16384 + 24576 + 128 + 32768 + 640;          // 74496
  lstm_recurrence_kernel<<<B_ / 16, 512, smem_bytes, stream>>>(
      x, Wbf, gpre, Wph, bp, (float*)d_out);
}
